// GoR_31782757990826
// MI455X (gfx1250) — matrix operand order repaired and checked by reading, not compiled, not run on hardware
//
#include <hip/hip_runtime.h>
#include <hip/hip_bf16.h>

// ---------------- problem constants ----------------
#define NN    8192      // nodes
#define DD    768       // input dim
#define HH    4         // heads
#define FF    768       // per-head feat
#define HDIM  3072      // H*F
#define DEGN  8         // edges per dst node
#define BGN   8         // graphs
#define NPGN  1024      // nodes per graph
#define QN    16        // queries per graph
#define QT    128       // BGN*QN

// ---------------- bf16 helpers ----------------
__device__ __forceinline__ unsigned short f2bf(float f) {
    unsigned int u = __float_as_uint(f);
    unsigned int r = (u + 0x7FFFu + ((u >> 16) & 1u)) >> 16;  // RNE
    return (unsigned short)r;
}
__device__ __forceinline__ float bf2f(unsigned short b) {
    return __uint_as_float(((unsigned int)b) << 16);
}

typedef __attribute__((ext_vector_type(16))) __bf16 v16bf;
typedef __attribute__((ext_vector_type(8)))  float  v8f;
struct B32x8 { uint4 lo, hi; };   // 32 bytes -> v16bf

// ---------------- generic f32 -> bf16 convert ----------------
__global__ void k_cvt_bf16(const float* __restrict__ src,
                           unsigned short* __restrict__ dst, int count) {
    int i = blockIdx.x * 256 + threadIdx.x;
    if (i < count) dst[i] = f2bf(src[i]);
}

// ---------------- WMMA GEMM: C[M,N] = A[M,K] @ B[K,N]  (bf16 in, f32 acc) ---
// Block: 256 threads = 8 waves arranged 4(M) x 2(N); each wave does a 32x64
// sub-tile as 2x4 16x16 WMMA tiles -> block tile 128(M) x 128(N); K step 32.
// A-fragments are loaded once per K-step and reused across the 4 N sub-tiles.
#define GTM 128
#define GTN 128
#define GTK 32
#define LDA_P 40
#define LDB_P 40

__global__ __launch_bounds__(256)
void k_gemm_bf16(const unsigned short* __restrict__ A,   // [M,K] bf16
                 const unsigned short* __restrict__ B,   // [K,N] bf16
                 float* __restrict__ Cf,                  // [M,N] f32 (or null)
                 unsigned short* __restrict__ Cb,         // [M,N] bf16 (or null)
                 int M, int N, int K)
{
    __shared__ __align__(16) unsigned short As[GTM * LDA_P];  // 10240 B
    __shared__ __align__(16) unsigned short Bs[GTN * LDB_P];  // transposed: Bs[n*LDB_P + k]

    const int tid  = threadIdx.x;
    const int lane = tid & 31;
    const int wave = tid >> 5;
    const int wm = wave >> 1;      // 0..3  (32-row band)
    const int wn = wave & 1;       // 0..1  (64-col band)
    const int mblk = blockIdx.y * GTM;
    const int nblk = blockIdx.x * GTN;

    v8f acc[2][4];
    #pragma unroll
    for (int i = 0; i < 2; ++i)
        #pragma unroll
        for (int j = 0; j < 4; ++j) { v8f z = {}; acc[i][j] = z; }

    // A-tile staging: 128 rows x 32 cols; thread -> (row, 16-elem segment)
    const int a_row = tid >> 1;
    const int a_col = (tid & 1) * 16;
    // B-tile staging: 32 rows x 128 cols; thread -> (k row, 16 n's), coalesced
    const int b_k  = tid >> 3;          // 0..31
    const int b_n0 = (tid & 7) * 16;    // 0,16,...,112

    const int r  = lane & 15;
    const int hi = lane >> 4;      // 0 or 1
    const int ka0 = hi * 8;        // A frag: elems 0..7 = K ka0..+7, 8..15 = +16
    const int kb0 = hi * 8;        // B frag: elems 0..7 = K kb0..+7, 8..15 = +16

    for (int kt = 0; kt < K; kt += GTK) {
        // prefetch next K-tile while this one is consumed
        if (kt + GTK < K) {
            __builtin_prefetch(A + (size_t)(mblk + a_row) * K + (kt + GTK) + a_col, 0, 3);
            __builtin_prefetch(B + (size_t)(kt + GTK + b_k) * N + nblk + b_n0, 0, 3);
        }
        // stage A (two 16B vector loads/stores per thread)
        const uint4* gA = (const uint4*)(A + (size_t)(mblk + a_row) * K + kt + a_col);
        uint4 av0 = gA[0];
        uint4 av1 = gA[1];
        *(uint4*)(&As[a_row * LDA_P + a_col])     = av0;
        *(uint4*)(&As[a_row * LDA_P + a_col + 8]) = av1;
        // stage B transposed (coalesced 32B global load, scalar LDS scatter)
        const uint4* gB = (const uint4*)(B + (size_t)(kt + b_k) * N + nblk + b_n0);
        uint4 bv0 = gB[0];
        uint4 bv1 = gB[1];
        const unsigned short* bvs0 = (const unsigned short*)&bv0;
        const unsigned short* bvs1 = (const unsigned short*)&bv1;
        #pragma unroll
        for (int i = 0; i < 8; ++i) Bs[(b_n0 + i) * LDB_P + b_k]     = bvs0[i];
        #pragma unroll
        for (int i = 0; i < 8; ++i) Bs[(b_n0 + 8 + i) * LDB_P + b_k] = bvs1[i];
        __syncthreads();

        // A fragments (loaded once, reused for all 4 N sub-tiles)
        v16bf afrag[2];
        #pragma unroll
        for (int i = 0; i < 2; ++i) {
            const int m0 = wm * 32 + i * 16;
            B32x8 ap;
            ap.lo = *(const uint4*)(&As[(m0 + r) * LDA_P + ka0]);
            ap.hi = *(const uint4*)(&As[(m0 + r) * LDA_P + ka0 + 16]);
            afrag[i] = __builtin_bit_cast(v16bf, ap);
        }
        #pragma unroll
        for (int j = 0; j < 4; ++j) {
            const int n0 = wn * 64 + j * 16;
            B32x8 bp;
            bp.lo = *(const uint4*)(&Bs[(n0 + r) * LDB_P + kb0]);
            bp.hi = *(const uint4*)(&Bs[(n0 + r) * LDB_P + kb0 + 16]);
            v16bf bfrag = __builtin_bit_cast(v16bf, bp);
            #pragma unroll
            for (int i = 0; i < 2; ++i) {
                acc[i][j] = __builtin_amdgcn_wmma_f32_16x16x32_bf16(
                    false, afrag[i], false, bfrag, (short)0, acc[i][j], false, false);
            }
        }
        __syncthreads();
    }

    // C/D layout: VGPR v -> M = v + 8*hi within tile, N = lane%16
    #pragma unroll
    for (int i = 0; i < 2; ++i) {
        const int mbase = mblk + wm * 32 + i * 16 + hi * 8;
        #pragma unroll
        for (int j = 0; j < 4; ++j) {
            const int n = nblk + wn * 64 + j * 16 + r;
            #pragma unroll
            for (int v = 0; v < 8; ++v) {
                float val = acc[i][j][v];
                size_t idx = (size_t)(mbase + v) * N + n;
                if (Cf) Cf[idx] = val;
                if (Cb) Cb[idx] = f2bf(val);
            }
        }
    }
}

// ---------------- el/er: per-node per-head attention dots ----------------
__global__ __launch_bounds__(256)
void k_elr(const unsigned short* __restrict__ featb,   // [NN, HDIM] bf16
           const float* __restrict__ al,               // [HH*FF]
           const float* __restrict__ ar,               // [HH*FF]
           float* __restrict__ el, float* __restrict__ er)  // [NN, HH]
{
    const int n = blockIdx.x;
    const int tid = threadIdx.x;
    const int grp = tid >> 5;          // 0..7 = (head, el/er)
    const int lane = tid & 31;
    const int h = grp >> 1;
    const int which = grp & 1;
    const float* w = which ? ar : al;
    const unsigned short* row = featb + (size_t)n * HDIM + h * FF;
    float s = 0.f;
    for (int f = lane; f < FF; f += 32) s += bf2f(row[f]) * w[h * FF + f];
    __shared__ float red[8][32];
    red[grp][lane] = s;
    __syncthreads();
    if (lane == 0) {
        float t = 0.f;
        #pragma unroll
        for (int i = 0; i < 32; ++i) t += red[grp][i];
        (which ? er : el)[n * HH + h] = t;
    }
}

// ---------------- edge softmax + weighted aggregation (+bias) -------------
__global__ __launch_bounds__(256)
void k_agg(const unsigned short* __restrict__ featb,  // [NN, HDIM] bf16
           const int* __restrict__ esrc,              // [NN*DEGN]
           const float* __restrict__ el, const float* __restrict__ er, // [NN,HH]
           const float* __restrict__ bias,            // [HDIM]
           float* __restrict__ out)                   // [NN, HDIM] f32
{
    const int n = blockIdx.x;
    const int tid = threadIdx.x;
    __shared__ float alpha[HH][DEGN];
    __shared__ float ev[HH][DEGN];
    __shared__ int srcs[DEGN];
    if (tid < DEGN) srcs[tid] = esrc[n * DEGN + tid];
    __syncthreads();
    if (tid < HH * DEGN) {
        const int h = tid / DEGN, e = tid % DEGN;
        float v = el[srcs[e] * HH + h] + er[n * HH + h];
        ev[h][e] = v > 0.f ? v : 0.2f * v;       // leaky relu
    }
    __syncthreads();
    if (tid < HH) {
        float m = -1e30f;
        #pragma unroll
        for (int e = 0; e < DEGN; ++e) m = fmaxf(m, ev[tid][e]);
        float d = 0.f;
        #pragma unroll
        for (int e = 0; e < DEGN; ++e) { float a = expf(ev[tid][e] - m); alpha[tid][e] = a; d += a; }
        const float inv = 1.f / d;
        #pragma unroll
        for (int e = 0; e < DEGN; ++e) alpha[tid][e] *= inv;
    }
    __syncthreads();
    for (int c = tid; c < HDIM; c += 256) {
        const int h = c / FF;
        float acc = bias[c];
        #pragma unroll
        for (int e = 0; e < DEGN; ++e)
            acc += alpha[h][e] * bf2f(featb[(size_t)srcs[e] * HDIM + c]);
        out[(size_t)n * HDIM + c] = acc;
    }
}

// ---------------- per-column mean / rsqrt(var+eps) over NN rows -----------
__global__ __launch_bounds__(256)
void k_colstats(const float* __restrict__ hf, float* __restrict__ mu,
                float* __restrict__ rvar)
{
    const int c = blockIdx.x;
    const int tid = threadIdx.x;
    float s = 0.f, s2 = 0.f;
    for (int r = tid; r < NN; r += 256) {
        float v = hf[(size_t)r * HDIM + c];
        s += v; s2 += v * v;
    }
    __shared__ float rs[256], rs2[256];
    rs[tid] = s; rs2[tid] = s2;
    __syncthreads();
    for (int o = 128; o > 0; o >>= 1) {
        if (tid < o) { rs[tid] += rs[tid + o]; rs2[tid] += rs2[tid + o]; }
        __syncthreads();
    }
    if (tid == 0) {
        float m = rs[0] / (float)NN;
        float v = rs2[0] / (float)NN - m * m;
        mu[c] = m;
        rvar[c] = rsqrtf(v + 1e-5f);
    }
}

// ---------------- batchnorm + prelu -> bf16 ----------------
__global__ void k_bn_prelu(const float* __restrict__ hf,
                           const float* __restrict__ mu, const float* __restrict__ rvar,
                           const float* __restrict__ gamma, const float* __restrict__ beta,
                           const float* __restrict__ pa,
                           unsigned short* __restrict__ outb)
{
    size_t i = (size_t)blockIdx.x * 256 + threadIdx.x;
    if (i >= (size_t)NN * HDIM) return;
    const int c = (int)(i % HDIM);
    const float x = hf[i];
    const float hn = (x - mu[c]) * rvar[c] * gamma[c] + beta[c];
    outb[i] = f2bf(hn > 0.f ? hn : pa[c] * hn);
}

// ---------------- node_rep = max(headmean(h1), headmean(h2)), transposed --
__global__ void k_noderep(const unsigned short* __restrict__ h1b,
                          const unsigned short* __restrict__ h2b,
                          unsigned short* __restrict__ nrT)   // [FF, NN] bf16
{
    int i = blockIdx.x * 256 + threadIdx.x;
    if (i >= NN * FF) return;
    const int n = i / FF, f = i % FF;
    float m1 = 0.f, m2 = 0.f;
    #pragma unroll
    for (int h = 0; h < HH; ++h) {
        m1 += bf2f(h1b[(size_t)n * HDIM + h * FF + f]);
        m2 += bf2f(h2b[(size_t)n * HDIM + h * FF + f]);
    }
    nrT[(size_t)f * NN + n] = f2bf(fmaxf(m1 * 0.25f, m2 * 0.25f));
}

// ---------------- bert-score descending rank per (g,q) row ----------------
__global__ __launch_bounds__(256)
void k_rank(const float* __restrict__ bs, int* __restrict__ perm)
{
    const int row = blockIdx.x;           // 0..QT-1
    const int tid = threadIdx.x;
    __shared__ float b[NPGN];
    for (int k = tid; k < NPGN; k += 256) b[k] = bs[(size_t)row * NPGN + k];
    __syncthreads();
    for (int k = tid; k < NPGN; k += 256) {
        const float v = b[k];
        int rank = 0;
        for (int j = 0; j < NPGN; ++j) {
            const float u = b[j];
            rank += (u > v) || (u == v && j < k);
        }
        perm[(size_t)row * NPGN + rank] = k;
    }
}

// ---------------- lambda-MRR pairwise loss sum per row --------------------
__global__ __launch_bounds__(256)
void k_mrr(const float* __restrict__ S, const int* __restrict__ perm,
           float* __restrict__ mrr_row)
{
    const int row = blockIdx.x;
    const int g = row / QN;
    const int tid = threadIdx.x;
    __shared__ float s[NPGN];
    for (int k = tid; k < NPGN; k += 256)
        s[k] = S[(size_t)row * NN + g * NPGN + perm[(size_t)row * NPGN + k]];
    __syncthreads();
    float acc = 0.f;
    for (int k = tid; k < NPGN - 1; k += 256) {
        const float sk = s[k];
        for (int l = k + 1; l < NPGN; ++l) {
            float d = sk - s[l];
            d = fminf(fmaxf(d, -50.f), 50.f);
            acc += log1pf(expf(-d));
        }
    }
    __shared__ float red[256];
    red[tid] = acc;
    __syncthreads();
    for (int o = 128; o > 0; o >>= 1) {
        if (tid < o) red[tid] += red[tid + o];
        __syncthreads();
    }
    if (tid == 0) mrr_row[row] = red[0];
}

// ---------------- contrastive + entropy per row ----------------
__global__ __launch_bounds__(256)
void k_cl_ent(const float* __restrict__ S, const int* __restrict__ perm,
              float* __restrict__ cl_row, float* __restrict__ ent_row)
{
    const int row = blockIdx.x;
    const int g = row / QN;
    const int tid = threadIdx.x;
    const float* r = S + (size_t)row * NN;
    const float* own = r + g * NPGN;
    __shared__ float red[256];

    float se = 0.f;
    for (int n = tid; n < NN; n += 256) se += expf(r[n]);
    float mx = -1e30f;
    for (int j = tid; j < NPGN; j += 256) mx = fmaxf(mx, own[j]);

    red[tid] = mx; __syncthreads();
    for (int o = 128; o > 0; o >>= 1) { if (tid < o) red[tid] = fmaxf(red[tid], red[tid + o]); __syncthreads(); }
    mx = red[0]; __syncthreads();

    float Z = 0.f, W = 0.f;
    for (int j = tid; j < NPGN; j += 256) {
        const float o = own[j];
        const float e = expf(o - mx);
        Z += e; W += o * e;
    }
    red[tid] = se; __syncthreads();
    for (int o = 128; o > 0; o >>= 1) { if (tid < o) red[tid] += red[tid + o]; __syncthreads(); }
    const float seT = red[0]; __syncthreads();
    red[tid] = Z; __syncthreads();
    for (int o = 128; o > 0; o >>= 1) { if (tid < o) red[tid] += red[tid + o]; __syncthreads(); }
    const float ZT = red[0]; __syncthreads();
    red[tid] = W; __syncthreads();
    for (int o = 128; o > 0; o >>= 1) { if (tid < o) red[tid] += red[tid + o]; __syncthreads(); }
    const float WT = red[0];

    if (tid == 0) {
        const float p_sim = own[perm[(size_t)row * NPGN + 0]];
        cl_row[row]  = logf(seT) - p_sim;        // -log(pe/(pe+ne)) = logsumexp_all - p_sim
        ent_row[row] = mx + logf(ZT) - WT / ZT;  // entropy of softmax over own graph
    }
}

// ---------------- final 3-scalar combine ----------------
__global__ void k_final(const float* __restrict__ cl_row,
                        const float* __restrict__ mrr_row,
                        const float* __restrict__ ent_row,
                        float* __restrict__ out)
{
    if (threadIdx.x == 0 && blockIdx.x == 0) {
        float c = 0.f, m = 0.f, e = 0.f;
        for (int i = 0; i < QT; ++i) { c += cl_row[i]; m += mrr_row[i]; e += ent_row[i]; }
        const float P = 0.5f * (float)NPGN * (float)(NPGN - 1);   // pairs with yt_i > yt_j per query
        out[0] = c / (float)QT;
        out[1] = m / ((float)QT * P);
        out[2] = e / (float)QT;
    }
}

// =================== host-side launcher ===================
extern "C" void kernel_launch(void* const* d_in, const int* in_sizes, int n_in,
                              void* d_out, int out_size, void* d_ws, size_t ws_size,
                              hipStream_t stream)
{
    const float* x      = (const float*)d_in[0];
    const int*   esrc   = (const int*)d_in[1];
    // d_in[2] = edge_dst: structurally repeat(arange(N), DEG) -> implicit
    const float* qemb   = (const float*)d_in[3];    // [BG,Q,D] = [128,768]
    const float* bscore = (const float*)d_in[4];    // [BG,Q,NPG]
    const float* W1     = (const float*)d_in[5];
    const float* al1    = (const float*)d_in[6];
    const float* ar1    = (const float*)d_in[7];
    const float* b1     = (const float*)d_in[8];
    const float* g1     = (const float*)d_in[9];
    const float* be1    = (const float*)d_in[10];
    const float* p1     = (const float*)d_in[11];
    const float* W2     = (const float*)d_in[12];
    const float* al2    = (const float*)d_in[13];
    const float* ar2    = (const float*)d_in[14];
    const float* b2     = (const float*)d_in[15];
    const float* g2     = (const float*)d_in[16];
    const float* be2    = (const float*)d_in[17];
    const float* p2     = (const float*)d_in[18];
    float* out = (float*)d_out;

    // ---- workspace carve-up (aligned 256B) ----
    size_t off = 0;
    auto alloc = [&](size_t bytes) -> void* {
        off = (off + 255) & ~(size_t)255;
        void* p = (char*)d_ws + off;
        off += bytes;
        return p;
    };
    unsigned short* xb    = (unsigned short*)alloc((size_t)NN * DD * 2);
    unsigned short* W1b   = (unsigned short*)alloc((size_t)DD * HDIM * 2);
    unsigned short* W2b   = (unsigned short*)alloc((size_t)HDIM * HDIM * 2);
    unsigned short* Qb    = (unsigned short*)alloc((size_t)QT * DD * 2);
    unsigned short* FEATb = (unsigned short*)alloc((size_t)NN * HDIM * 2);
    float*          HF    = (float*)alloc((size_t)NN * HDIM * 4);
    unsigned short* H1b   = (unsigned short*)alloc((size_t)NN * HDIM * 2);
    unsigned short* H2b   = (unsigned short*)alloc((size_t)NN * HDIM * 2);
    unsigned short* NRTb  = (unsigned short*)alloc((size_t)FF * NN * 2);
    float*          Smat  = (float*)alloc((size_t)QT * NN * 4);
    float*          el    = (float*)alloc((size_t)NN * HH * 4);
    float*          er    = (float*)alloc((size_t)NN * HH * 4);
    float*          mu    = (float*)alloc((size_t)HDIM * 4);
    float*          rvar  = (float*)alloc((size_t)HDIM * 4);
    int*            perm  = (int*)alloc((size_t)QT * NPGN * 4);
    float*          clr   = (float*)alloc((size_t)QT * 4);
    float*          mrrr  = (float*)alloc((size_t)QT * 4);
    float*          entr  = (float*)alloc((size_t)QT * 4);
    (void)ws_size; (void)in_sizes; (void)n_in; (void)out_size;

    // ---- converts ----
    k_cvt_bf16<<<(NN * DD + 255) / 256, 256, 0, stream>>>(x, xb, NN * DD);
    k_cvt_bf16<<<(DD * HDIM + 255) / 256, 256, 0, stream>>>(W1, W1b, DD * HDIM);
    k_cvt_bf16<<<(HDIM * HDIM + 255) / 256, 256, 0, stream>>>(W2, W2b, HDIM * HDIM);
    k_cvt_bf16<<<(QT * DD + 255) / 256, 256, 0, stream>>>(qemb, Qb, QT * DD);

    const int bnBlocks = (int)(((size_t)NN * HDIM + 255) / 256);

    // ---- GAT layer 1 ----
    k_gemm_bf16<<<dim3(HDIM / GTN, NN / GTM), 256, 0, stream>>>(
        xb, W1b, nullptr, FEATb, NN, HDIM, DD);
    k_elr<<<NN, 256, 0, stream>>>(FEATb, al1, ar1, el, er);
    k_agg<<<NN, 256, 0, stream>>>(FEATb, esrc, el, er, b1, HF);
    k_colstats<<<HDIM, 256, 0, stream>>>(HF, mu, rvar);
    k_bn_prelu<<<bnBlocks, 256, 0, stream>>>(HF, mu, rvar, g1, be1, p1, H1b);

    // ---- GAT layer 2 ----
    k_gemm_bf16<<<dim3(HDIM / GTN, NN / GTM), 256, 0, stream>>>(
        H1b, W2b, nullptr, FEATb, NN, HDIM, HDIM);
    k_elr<<<NN, 256, 0, stream>>>(FEATb, al2, ar2, el, er);
    k_agg<<<NN, 256, 0, stream>>>(FEATb, esrc, el, er, b2, HF);
    k_colstats<<<HDIM, 256, 0, stream>>>(HF, mu, rvar);
    k_bn_prelu<<<bnBlocks, 256, 0, stream>>>(HF, mu, rvar, g2, be2, p2, H2b);

    // ---- node representations (transposed bf16) + similarity GEMM ----
    k_noderep<<<(NN * FF + 255) / 256, 256, 0, stream>>>(H1b, H2b, NRTb);
    k_gemm_bf16<<<dim3(NN / GTN, QT / GTM), 256, 0, stream>>>(
        Qb, NRTb, Smat, nullptr, QT, NN, DD);

    // ---- losses ----
    k_rank<<<QT, 256, 0, stream>>>(bscore, perm);
    k_mrr<<<QT, 256, 0, stream>>>(Smat, perm, mrrr);
    k_cl_ent<<<QT, 256, 0, stream>>>(Smat, perm, clr, entr);
    k_final<<<1, 64, 0, stream>>>(clr, mrrr, entr, out);
}
